// MambaLayer_67078799229628
// MI455X (gfx1250) — compile-verified, matrix operand order checked by reading, not run on hardware
//
#include <hip/hip_runtime.h>
#include <hip/hip_bf16.h>
#include <math.h>

#define DIM     768
#define DINNER  1536
#define DSTATE  64
#define SEQL    1024
#define NBATCH  2
#define DTRANK  48
#define PROJP   256                     /* padded proj width: 48 dt |16 z| 64 B | 64 C | 64 z */
#define MTOT    (NBATCH * SEQL)         /* 2048 */

typedef __attribute__((ext_vector_type(16))) __bf16        v16bf;
typedef __attribute__((ext_vector_type(8)))  float         v8f;
typedef __attribute__((ext_vector_type(4)))  float         v4f;
typedef __attribute__((ext_vector_type(4)))  unsigned int  v4u;
typedef __attribute__((ext_vector_type(2)))  unsigned int  v2u;

__device__ __forceinline__ v2u cvt4bf16(v4f f) {
  union { __bf16 h[4]; v2u u; } r;
  r.h[0] = (__bf16)f.x; r.h[1] = (__bf16)f.y;
  r.h[2] = (__bf16)f.z; r.h[3] = (__bf16)f.w;
  return r.u;
}

// ---------------------------------------------------------------------------
// LayerNorm: one block per token row of 768.
// ---------------------------------------------------------------------------
__global__ __launch_bounds__(256)
void ln_kernel(const float* __restrict__ x, const float* __restrict__ g,
               const float* __restrict__ bta, float* __restrict__ xn)
{
  __shared__ float s1[256];
  __shared__ float s2[256];
  const int row = blockIdx.x;
  const int tid = threadIdx.x;
  const long base = (long)row * DIM;

  float a0 = x[base + tid];
  float a1 = x[base + tid + 256];
  float a2 = x[base + tid + 512];
  s1[tid] = a0 + a1 + a2;
  s2[tid] = a0 * a0 + a1 * a1 + a2 * a2;
  __syncthreads();
  for (int st = 128; st > 0; st >>= 1) {
    if (tid < st) { s1[tid] += s1[tid + st]; s2[tid] += s2[tid + st]; }
    __syncthreads();
  }
  const float mean = s1[0] * (1.0f / DIM);
  const float var  = s2[0] * (1.0f / DIM) - mean * mean;
  const float rstd = rsqrtf(var + 1e-5f);
  xn[base + tid      ] = (a0 - mean) * rstd * g[tid      ] + bta[tid      ];
  xn[base + tid + 256] = (a1 - mean) * rstd * g[tid + 256] + bta[tid + 256];
  xn[base + tid + 512] = (a2 - mean) * rstd * g[tid + 512] + bta[tid + 512];
}

// ---------------------------------------------------------------------------
// Weight padding (once per call):
//  xpw_pad (256 x 1536): rows 0..47 = x_proj_w dt rows, 48..63 = 0,
//                        64..191 = x_proj_w rows 48..175 (B|C), 192..255 = 0.
//  dtw_pad (1536 x 64):  cols 0..47 = dt_proj_w, cols 48..63 = 0.
// ---------------------------------------------------------------------------
__global__ __launch_bounds__(256)
void pad_xpw_kernel(const float* __restrict__ w, float* __restrict__ wp)
{
  const int idx = blockIdx.x * 256 + threadIdx.x;   // [0, 256*1536)
  const int n = idx / DINNER;
  const int k = idx % DINNER;
  float v = 0.0f;
  if (n < DTRANK)                 v = w[n * DINNER + k];
  else if (n >= 64 && n < 192)    v = w[(n - 16) * DINNER + k];
  wp[idx] = v;
}

__global__ __launch_bounds__(256)
void pad_dtw_kernel(const float* __restrict__ w, float* __restrict__ wp)
{
  const int idx = blockIdx.x * 256 + threadIdx.x;   // [0, 1536*64)
  const int n = idx >> 6;
  const int k = idx & 63;
  wp[idx] = (k < DTRANK) ? w[n * DTRANK + k] : 0.0f;
}

// ---------------------------------------------------------------------------
// C = A[*,K](row-major, stride lda) @ W[N,K]^T (row-major, stride K).
// fp32 in/out, bf16 WMMA, fp32 accumulate. Block tile 64(M) x 128(N),
// 8 waves as 2x4, each wave 32x32 = 2x2 WMMA tiles. All dims tile-divisible:
// fully branchless staging with register double-buffering of the next chunk.
// ---------------------------------------------------------------------------
template <bool SOFTPLUS_BIAS>
__global__ __launch_bounds__(256)
void gemm_bf16w(const float* __restrict__ A, const float* __restrict__ W,
                float* __restrict__ C, const float* __restrict__ bias,
                int K, int lda, int ldc)
{
  // row stride 40 u16 = 80B: 16B-aligned rows, conflict-mitigated frag reads.
  __shared__ __align__(16) unsigned short lsA[64 * 40];
  __shared__ __align__(16) unsigned short lsB[128 * 40];

  const int tid   = threadIdx.x;
  const int lane  = tid & 31;
  const int wave  = tid >> 5;
  const int waveM = wave >> 2;   // 0..1
  const int waveN = wave & 3;    // 0..3
  const int m16   = lane & 15;
  const int half  = lane >> 4;

  const int blockN = blockIdx.x * 128;
  const int blockM = blockIdx.y * 64;

  // Per-thread staging slots (float4 granularity).
  const float* aG[2]; int aO[2];
  const float* bG[4]; int bO[4];
#pragma unroll
  for (int i = 0; i < 2; ++i) {
    const int q = tid + i * 256;        // 0..511 : 64 rows x 8 float4
    const int r = q >> 3, c4 = q & 7;
    aG[i] = A + (long)(blockM + r) * lda + c4 * 4;
    aO[i] = r * 40 + c4 * 4;
  }
#pragma unroll
  for (int i = 0; i < 4; ++i) {
    const int q = tid + i * 256;        // 0..1023 : 128 rows x 8 float4
    const int r = q >> 3, c4 = q & 7;
    bG[i] = W + (long)(blockN + r) * K + c4 * 4;
    bO[i] = r * 40 + c4 * 4;
  }

  v8f acc[2][2];
#pragma unroll
  for (int i = 0; i < 2; ++i)
#pragma unroll
    for (int j = 0; j < 2; ++j)
#pragma unroll
      for (int r = 0; r < 8; ++r) acc[i][j][r] = 0.0f;

  v4f aR[2], bR[4];
#pragma unroll
  for (int i = 0; i < 2; ++i) aR[i] = *(const v4f*)aG[i];
#pragma unroll
  for (int i = 0; i < 4; ++i) bR[i] = *(const v4f*)bG[i];

  const int kChunks = K >> 5;
  for (int kc = 0; kc < kChunks; ++kc) {
    // Commit current chunk to LDS (fp32 -> packed bf16, one b64 store each).
#pragma unroll
    for (int i = 0; i < 2; ++i) *(v2u*)&lsA[aO[i]] = cvt4bf16(aR[i]);
#pragma unroll
    for (int i = 0; i < 4; ++i) *(v2u*)&lsB[bO[i]] = cvt4bf16(bR[i]);
    __syncthreads();

    // Refill registers for next chunk (overlaps with WMMA below).
    if (kc + 1 < kChunks) {
      const int koff = (kc + 1) * 32;
#pragma unroll
      for (int i = 0; i < 2; ++i) aR[i] = *(const v4f*)(aG[i] + koff);
#pragma unroll
      for (int i = 0; i < 4; ++i) bR[i] = *(const v4f*)(bG[i] + koff);
      if (kc + 2 < kChunks) {       // deepen pipeline: global_prefetch
        __builtin_prefetch(aG[0] + koff + 32, 0, 0);
        __builtin_prefetch(bG[0] + koff + 32, 0, 0);
      }
    }

    // Fragment loads per documented 16-bit A/B layout:
    // lane(m16,half) reads K pairs at byte offsets half*16 and 32+half*16.
    v16bf aF[2], bF[2];
#pragma unroll
    for (int i = 0; i < 2; ++i) {
      const unsigned short* rp = &lsA[(waveM * 32 + i * 16 + m16) * 40];
      union { v4u u[2]; v16bf v; } cvt;
      cvt.u[0] = *(const v4u*)(rp + half * 8);
      cvt.u[1] = *(const v4u*)(rp + 16 + half * 8);
      aF[i] = cvt.v;
    }
#pragma unroll
    for (int j = 0; j < 2; ++j) {
      const unsigned short* rp = &lsB[(waveN * 32 + j * 16 + m16) * 40];
      union { v4u u[2]; v16bf v; } cvt;
      cvt.u[0] = *(const v4u*)(rp + half * 8);
      cvt.u[1] = *(const v4u*)(rp + 16 + half * 8);
      bF[j] = cvt.v;
    }

#pragma unroll
    for (int i = 0; i < 2; ++i)
#pragma unroll
      for (int j = 0; j < 2; ++j)
        acc[i][j] = __builtin_amdgcn_wmma_f32_16x16x32_bf16(
            false, aF[i], false, bF[j], (short)0, acc[i][j], false, false);
    __syncthreads();
  }

  // Epilogue: C/D layout = VGPR r holds (m = r + 8*half, n = m16).
#pragma unroll
  for (int i = 0; i < 2; ++i)
#pragma unroll
    for (int j = 0; j < 2; ++j) {
      const int gn = blockN + waveN * 32 + j * 16 + m16;
      float bval = SOFTPLUS_BIAS ? bias[gn] : 0.0f;
#pragma unroll
      for (int r = 0; r < 8; ++r) {
        const int gm = blockM + waveM * 32 + i * 16 + r + 8 * half;
        float v = acc[i][j][r];
        if (SOFTPLUS_BIAS) {
          v += bval;
          v = (v > 20.0f) ? v : log1pf(__expf(v));
        }
        C[(long)gm * ldc + gn] = v;
      }
    }
}

// ---------------------------------------------------------------------------
// Causal depthwise conv (kernel 4) + SiLU. xin = first half of xz rows.
// ---------------------------------------------------------------------------
__global__ __launch_bounds__(256)
void conv_silu_kernel(const float* __restrict__ xz, const float* __restrict__ cw,
                      const float* __restrict__ cb, float* __restrict__ xa)
{
  const int idx = blockIdx.x * 256 + threadIdx.x;   // [0, MTOT*DINNER)
  const int d   = idx % DINNER;
  const int row = idx / DINNER;
  const int l   = row & (SEQL - 1);
  float acc = cb[d];
#pragma unroll
  for (int k = 0; k < 4; ++k) {
    const int ls = l - 3 + k;
    if (ls >= 0) acc += xz[(long)(row - 3 + k) * (2 * DINNER) + d] * cw[d * 4 + k];
  }
  xa[(long)row * DINNER + d] = acc / (1.0f + __expf(-acc));
}

// ---------------------------------------------------------------------------
// Selective scan: one wave32 per (batch, channel). 64 SSM states = 2/lane.
// Fused epilogue: y = (sum_n h*C + x*D) * silu(z).
// proj layout (padded): [0,48) dt_raw | [64,128) B | [128,192) C, stride 256.
// ---------------------------------------------------------------------------
__global__ __launch_bounds__(256)
void scan_kernel(const float* __restrict__ dt, const float* __restrict__ xa,
                 const float* __restrict__ proj, const float* __restrict__ xz,
                 const float* __restrict__ A_log, const float* __restrict__ Dv,
                 float* __restrict__ yf)
{
  const int lane = threadIdx.x & 31;
  const int wave = threadIdx.x >> 5;
  const int w    = blockIdx.x * 8 + wave;
  const int b    = w / DINNER;
  const int d    = w % DINNER;

  const float Aa = -__expf(A_log[d * DSTATE + lane]);
  const float Ab = -__expf(A_log[d * DSTATE + lane + 32]);
  const float Dd = Dv[d];

  float h0 = 0.0f, h1 = 0.0f;
  for (int t = 0; t < SEQL; ++t) {
    const long row   = (long)b * SEQL + t;
    const float dt_t = dt[row * DINNER + d];
    const float x_t  = xa[row * DINNER + d];
    const float Bn0  = proj[row * PROJP + 64 + lane];
    const float Bn1  = proj[row * PROJP + 64 + lane + 32];
    const float Cn0  = proj[row * PROJP + 128 + lane];
    const float Cn1  = proj[row * PROJP + 128 + lane + 32];

    h0 = __expf(dt_t * Aa) * h0 + dt_t * Bn0 * x_t;
    h1 = __expf(dt_t * Ab) * h1 + dt_t * Bn1 * x_t;

    float p = h0 * Cn0 + h1 * Cn1;
    p += __shfl_xor(p, 16, 32);
    p += __shfl_xor(p,  8, 32);
    p += __shfl_xor(p,  4, 32);
    p += __shfl_xor(p,  2, 32);
    p += __shfl_xor(p,  1, 32);

    if (lane == 0) {
      const float z  = xz[row * (2 * DINNER) + DINNER + d];
      const float sz = z / (1.0f + __expf(-z));
      yf[row * DINNER + d] = (p + x_t * Dd) * sz;
    }
  }
}

// ---------------------------------------------------------------------------
extern "C" void kernel_launch(void* const* d_in, const int* in_sizes, int n_in,
                              void* d_out, int out_size, void* d_ws, size_t ws_size,
                              hipStream_t stream) {
  (void)in_sizes; (void)n_in; (void)out_size; (void)ws_size;

  const float* x          = (const float*)d_in[0];
  const float* ln_gamma   = (const float*)d_in[1];
  const float* ln_beta    = (const float*)d_in[2];
  const float* in_proj_w  = (const float*)d_in[3];
  const float* conv_w     = (const float*)d_in[4];
  const float* conv_b     = (const float*)d_in[5];
  const float* x_proj_w   = (const float*)d_in[6];
  const float* dt_proj_w  = (const float*)d_in[7];
  const float* dt_proj_b  = (const float*)d_in[8];
  const float* A_log      = (const float*)d_in[9];
  const float* Dvec       = (const float*)d_in[10];
  const float* out_proj_w = (const float*)d_in[11];
  float*       out        = (float*)d_out;

  float* ws      = (float*)d_ws;
  float* xn      = ws;                              // 2048 x 768
  float* xz      = xn      + (long)MTOT * DIM;      // 2048 x 3072
  float* xa      = xz      + (long)MTOT * 2 * DINNER; // 2048 x 1536
  float* projp   = xa      + (long)MTOT * DINNER;   // 2048 x 256 (padded)
  float* dt      = projp   + (long)MTOT * PROJP;    // 2048 x 1536
  float* yf      = dt      + (long)MTOT * DINNER;   // 2048 x 1536
  float* xpw_pad = yf      + (long)MTOT * DINNER;   // 256 x 1536
  float* dtw_pad = xpw_pad + (long)PROJP * DINNER;  // 1536 x 64

  // 0) pad weights (zero-filled ragged dims -> branchless GEMMs)
  pad_xpw_kernel<<<(PROJP * DINNER) / 256, 256, 0, stream>>>(x_proj_w, xpw_pad);
  pad_dtw_kernel<<<(DINNER * 64) / 256, 256, 0, stream>>>(dt_proj_w, dtw_pad);

  // 1) LayerNorm
  ln_kernel<<<MTOT, 256, 0, stream>>>(x, ln_gamma, ln_beta, xn);

  // 2) in_proj: xz = xn @ in_proj_w^T   (M=2048, N=3072, K=768)
  gemm_bf16w<false><<<dim3((2 * DINNER) / 128, MTOT / 64), 256, 0, stream>>>(
      xn, in_proj_w, xz, nullptr, DIM, DIM, 2 * DINNER);

  // 3) causal depthwise conv + SiLU
  conv_silu_kernel<<<(MTOT * DINNER) / 256, 256, 0, stream>>>(xz, conv_w, conv_b, xa);

  // 4) x_proj: projp = xa @ xpw_pad^T   (M=2048, N=256, K=1536)
  gemm_bf16w<false><<<dim3(PROJP / 128, MTOT / 64), 256, 0, stream>>>(
      xa, xpw_pad, projp, nullptr, DINNER, DINNER, PROJP);

  // 5) dt_proj + softplus: dt = softplus(projp[:, :64] @ dtw_pad^T + b)
  //    (cols 48..63 of projp are exact zeros -> equivalent to K=48)
  gemm_bf16w<true><<<dim3(DINNER / 128, MTOT / 64), 256, 0, stream>>>(
      projp, dtw_pad, dt, dt_proj_b, 64, PROJP, DINNER);

  // 6) selective scan + gating epilogue
  scan_kernel<<<(NBATCH * DINNER) / 8, 256, 0, stream>>>(dt, xa, projp, xz, A_log, Dvec, yf);

  // 7) out_proj: out = yf @ out_proj_w^T  (M=2048, N=768, K=1536)
  gemm_bf16w<false><<<dim3(DIM / 128, MTOT / 64), 256, 0, stream>>>(
      yf, out_proj_w, out, nullptr, DINNER, DINNER, DIM);
}
